// FastBEV_18442589569668
// MI455X (gfx1250) — matrix operand order repaired and checked by reading, not compiled, not run on hardware
//
#include <hip/hip_runtime.h>
#include <hip/hip_bf16.h>
#include <math.h>

typedef __attribute__((ext_vector_type(8)))  _Float16 v8h;
typedef __attribute__((ext_vector_type(16))) _Float16 v16h;
typedef __attribute__((ext_vector_type(8)))  float    v8f;

// ---------------------------------------------------------------------------
// Fold conv bias + BN into per-channel scale/bias:  y = conv*scale + bias
// ---------------------------------------------------------------------------
__global__ void bnfold_kernel(const float* __restrict__ bconv,
                              const float* __restrict__ g,
                              const float* __restrict__ beta,
                              const float* __restrict__ mean,
                              const float* __restrict__ var,
                              float* __restrict__ scale,
                              float* __restrict__ bias, int C)
{
    int c = blockIdx.x * blockDim.x + threadIdx.x;
    if (c >= C) return;
    float s = g[c] / sqrtf(var[c] + 1e-5f);
    scale[c] = s;
    bias[c]  = (bconv[c] - mean[c]) * s + beta[c];
}

// ---------------------------------------------------------------------------
// OIHW f32 -> [tap][O][I] f16 weight transform (K = input channel contiguous)
// ---------------------------------------------------------------------------
__global__ void wxform_kernel(const float* __restrict__ w, _Float16* __restrict__ wt,
                              int Cout, int Cin)
{
    int idx = blockIdx.x * blockDim.x + threadIdx.x;
    int total = 9 * Cout * Cin;
    if (idx >= total) return;
    int i   = idx % Cin;
    int t   = idx / Cin;
    int o   = t % Cout;
    int tap = t / Cout;
    int kh = tap / 3, kw = tap % 3;
    wt[idx] = (_Float16)w[(((size_t)o * Cin + i) * 3 + kh) * 3 + kw];
}

__global__ void zerofill_kernel(_Float16* __restrict__ p, int n)
{
    int i = blockIdx.x * blockDim.x + threadIdx.x;
    if (i < n) p[i] = (_Float16)0.f;
}

// ---------------------------------------------------------------------------
// First conv: 3 -> 32, 3x3 stride 2, ReLU. NCHW f32 in, NHWC f16 out.
// ---------------------------------------------------------------------------
__global__ void conv1_direct(const float* __restrict__ imgs,
                             const float* __restrict__ w,
                             const float* __restrict__ bias,
                             _Float16* __restrict__ out)
{
    const int Hin = 256, Win = 704, Hout = 128, Wout = 352;
    const int Npix = 6 * Hout * Wout;
    int p = blockIdx.x * blockDim.x + threadIdx.x;
    if (p >= Npix) return;
    int img = p / (Hout * Wout);
    int rem = p - img * (Hout * Wout);
    int oy = rem / Wout;
    int ox = rem - oy * Wout;
    const float* base = imgs + (size_t)img * 3 * Hin * Win;
    float patch[27];
#pragma unroll
    for (int c = 0; c < 3; ++c)
#pragma unroll
        for (int kh = 0; kh < 3; ++kh)
#pragma unroll
            for (int kw = 0; kw < 3; ++kw) {
                int iy = oy * 2 + kh - 1, ix = ox * 2 + kw - 1;
                float val = 0.f;
                if (iy >= 0 && iy < Hin && ix >= 0 && ix < Win)
                    val = base[(size_t)c * Hin * Win + (size_t)iy * Win + ix];
                patch[(c * 3 + kh) * 3 + kw] = val;
            }
    _Float16* op = out + (size_t)p * 32;
    for (int oc = 0; oc < 32; ++oc) {
        float acc = bias[oc];
        const float* wp = w + oc * 27;
#pragma unroll
        for (int k = 0; k < 27; ++k) acc += patch[k] * wp[k];
        acc = acc > 0.f ? acc : 0.f;
        op[oc] = (_Float16)acc;
    }
}

// ---------------------------------------------------------------------------
// Simple per-wave WMMA conv (16 px x 16 oc per wave), branchless halo via
// zero page. Used for the small backbone convs (Cout 32/64).
// ---------------------------------------------------------------------------
__global__ __launch_bounds__(256)
void conv3x3_wmma(const _Float16* __restrict__ in, const _Float16* __restrict__ wt,
                  const float* __restrict__ scale, const float* __restrict__ bias,
                  const _Float16* __restrict__ zbuf,
                  _Float16* __restrict__ out,
                  int Nimg, int Hin, int Win, int Cin,
                  int Hout, int Wout, int Cout, int stride)
{
    const int wave = threadIdx.x >> 5;
    const int lane = threadIdx.x & 31;
    const int coTiles = Cout >> 4;
    const int totalPix = Nimg * Hout * Wout;
    const int ptiles = (totalPix + 15) >> 4;
    const int totalTiles = ptiles * coTiles;
    int tile = blockIdx.x * 8 + wave;
    if (tile >= totalTiles) return;               // wave-uniform
    int pt = tile / coTiles;
    int ot = tile - pt * coTiles;

    const int m = lane & 15;
    const int g = lane >> 4;
    int p = pt * 16 + m;
    bool pvalid = p < totalPix;
    int pc = pvalid ? p : 0;
    int img = pc / (Hout * Wout);
    int rem = pc - img * (Hout * Wout);
    int oy = rem / Wout;
    int ox = rem - oy * Wout;
    int obase = ot << 4;

    const _Float16* inImg = in + (size_t)img * Hin * Win * Cin;
    const int kblocks = Cin >> 5;

    v8f acc = {};
    for (int kh = 0; kh < 3; ++kh) {
        int iy = oy * stride + kh - 1;
        bool yok = (iy >= 0) & (iy < Hin);
        for (int kw = 0; kw < 3; ++kw) {
            int ix = ox * stride + kw - 1;
            bool ok = pvalid & yok & (ix >= 0) & (ix < Win);
            const _Float16* asrc = inImg + ((size_t)iy * Win + ix) * Cin;
            const _Float16* bsrc = wt + ((size_t)(kh * 3 + kw) * Cout + (obase + m)) * Cin;
            for (int kb = 0; kb < kblocks; ++kb) {
                int c0 = kb << 5;
                const _Float16* pa = ok ? (asrc + c0) : zbuf;   // branchless halo
                v8h a0 = *(const v8h*)(pa + g * 8);
                v8h a1 = *(const v8h*)(pa + 16 + g * 8);
                v16h a = __builtin_shufflevector(a0, a1,
                           0,1,2,3,4,5,6,7,8,9,10,11,12,13,14,15);
                v8h b0 = *(const v8h*)(bsrc + c0 + g * 8);
                v8h b1 = *(const v8h*)(bsrc + c0 + 16 + g * 8);
                v16h b = __builtin_shufflevector(b0, b1,
                           0,1,2,3,4,5,6,7,8,9,10,11,12,13,14,15);
                acc = __builtin_amdgcn_wmma_f32_16x16x32_f16(
                          false, a, false, b, (short)0, acc, false, false);
            }
        }
    }

    int n = obase + m;
    float sc = scale[n], bi = bias[n];
#pragma unroll
    for (int j = 0; j < 8; ++j) {
        int mm = j + (g << 3);
        int pp = pt * 16 + mm;
        if (pp < totalPix) {
            float vv = acc[j] * sc + bi;
            vv = vv > 0.f ? vv : 0.f;
            out[(size_t)pp * Cout + n] = (_Float16)vv;
        }
    }
}

// ---------------------------------------------------------------------------
// LDS-staged WMMA conv for Cout=128 (s3, e1, e2).
// Block = 256 threads = 8 waves computes a 64-pixel x 128-outch macro tile.
// Per 32-wide K-step: stage A (64x32 f16) + B (128x32 f16) in LDS (rows padded
// to 40 halves for bank-conflict-free fragment reads), double buffered, one
// barrier per step; each wave runs 4 WMMAs (16 px x 64 oc) reusing one A frag.
// Arithmetic intensity ~43 flops/byte of global traffic vs 8 for the naive
// scheme -> e1 drops from ~6.3 GB to ~1.2 GB of L2 traffic.
// ---------------------------------------------------------------------------
__global__ __launch_bounds__(256)
void conv3x3_wmma_lds(const _Float16* __restrict__ in, const _Float16* __restrict__ wt,
                      const float* __restrict__ scale, const float* __restrict__ bias,
                      const _Float16* __restrict__ zbuf,
                      _Float16* __restrict__ out,
                      int Nimg, int Hin, int Win, int Cin,
                      int Hout, int Wout, int stride)
{
    const int CO = 128;
    const int LDA = 40;                       // padded row stride (halves)
    __shared__ __align__(16) _Float16 As[2][64 * 40];
    __shared__ __align__(16) _Float16 Bs[2][128 * 40];

    const int tid  = threadIdx.x;
    const int wave = tid >> 5;
    const int lane = tid & 31;
    const int pb   = blockIdx.x;              // 64-pixel macro tile index

    // A-loader coords: thread t stages pixel apx, 8-half quarter aq
    const int apx = tid >> 2;
    const int aq  = tid & 3;
    int pA = pb * 64 + apx;
    int imgA = pA / (Hout * Wout);
    int remA = pA - imgA * (Hout * Wout);
    int oyA = remA / Wout;
    int oxA = remA - oyA * Wout;
    const _Float16* inImgA = in + (size_t)imgA * Hin * Win * Cin;

    // B-loader coords: thread t stages outch boc, 16-half half bh
    const int boc = tid >> 1;
    const int bh  = tid & 1;

    const int kblocks = Cin >> 5;
    const int steps = 9 * kblocks;

    // compute-phase coords
    const int m = lane & 15;
    const int g = lane >> 4;
    const int pxw = (wave >> 1) << 4;         // 0,16,32,48
    const int ocw = (wave & 1) << 6;          // 0,64

    v8f acc[4] = {{}, {}, {}, {}};

    v8h ra, rb0, rb1;
    auto fetch = [&](int s) {
        int tap = s / kblocks;
        int kb  = s - tap * kblocks;
        int kh = tap / 3, kw = tap - kh * 3;
        int iy = oyA * stride + kh - 1;
        int ix = oxA * stride + kw - 1;
        bool ok = (iy >= 0) & (iy < Hin) & (ix >= 0) & (ix < Win);
        const _Float16* pa = inImgA + ((size_t)iy * Win + ix) * Cin + (kb << 5) + aq * 8;
        ra = *(const v8h*)(ok ? pa : zbuf);
        const _Float16* pbm = wt + ((size_t)tap * CO + boc) * Cin + (kb << 5) + bh * 16;
        rb0 = *(const v8h*)pbm;
        rb1 = *(const v8h*)(pbm + 8);
    };
    auto stage = [&](int buf) {
        *(v8h*)&As[buf][apx * LDA + aq * 8] = ra;
        *(v8h*)&Bs[buf][boc * LDA + bh * 16] = rb0;
        *(v8h*)&Bs[buf][boc * LDA + bh * 16 + 8] = rb1;
    };

    fetch(0);
    stage(0);
    int cur = 0;
    for (int s = 0; s < steps; ++s) {
        __syncthreads();                      // staged data for step s visible
        bool more = (s + 1) < steps;
        if (more) fetch(s + 1);               // global loads hide under compute

        const _Float16* arow = &As[cur][(pxw + m) * LDA];
        v8h a0 = *(const v8h*)(arow + g * 8);
        v8h a1 = *(const v8h*)(arow + 16 + g * 8);
        v16h a = __builtin_shufflevector(a0, a1,
                   0,1,2,3,4,5,6,7,8,9,10,11,12,13,14,15);
#pragma unroll
        for (int f = 0; f < 4; ++f) {
            const _Float16* brow = &Bs[cur][(ocw + f * 16 + m) * LDA];
            v8h b0 = *(const v8h*)(brow + g * 8);
            v8h b1 = *(const v8h*)(brow + 16 + g * 8);
            v16h b = __builtin_shufflevector(b0, b1,
                       0,1,2,3,4,5,6,7,8,9,10,11,12,13,14,15);
            acc[f] = __builtin_amdgcn_wmma_f32_16x16x32_f16(
                         false, a, false, b, (short)0, acc[f], false, false);
        }
        if (more) stage(cur ^ 1);             // safe: barrier above fenced s-1 reads
        cur ^= 1;
    }

    int pbase = pb * 64 + pxw;
#pragma unroll
    for (int f = 0; f < 4; ++f) {
        int n = ocw + f * 16 + m;
        float sc = scale[n], bi = bias[n];
#pragma unroll
        for (int e = 0; e < 8; ++e) {
            int pp = pbase + e + (g << 3);
            float vv = acc[f][e] * sc + bi;
            vv = vv > 0.f ? vv : 0.f;
            out[(size_t)pp * CO + n] = (_Float16)vv;
        }
    }
}

// ---------------------------------------------------------------------------
// Per-voxel projection LUT for 3 scales (first-valid-camera priority).
// ---------------------------------------------------------------------------
__global__ void lut_kernel(const float* __restrict__ intr, const float* __restrict__ extr,
                           int* __restrict__ lut)
{
    const int NV = 128 * 128 * 6;
    int v = blockIdx.x * blockDim.x + threadIdx.x;
    if (v >= NV) return;
    int iz = v % 6;
    int t  = v / 6;
    int iy = t % 128;
    int ix = t / 128;
    float X = ix * 0.8f - 51.2f + 0.4f;
    float Y = iy * 0.8f - 51.2f + 0.4f;
    float Z = iz * 1.0f - 3.0f + 0.5f;

    const int   Hs[3] = {64, 32, 16};
    const int   Ws[3] = {176, 88, 44};
    const float Ss[3] = {4.f, 8.f, 16.f};
    int flat[3] = {0, 0, 0};
    int found[3] = {0, 0, 0};

    for (int cam = 0; cam < 6; ++cam) {
        const float* E = extr + cam * 16;
        const float* K = intr + cam * 9;
        float p0 = E[0] * X + E[1] * Y + E[2]  * Z + E[3];
        float p1 = E[4] * X + E[5] * Y + E[6]  * Z + E[7];
        float p2 = E[8] * X + E[9] * Y + E[10] * Z + E[11];
        float q0 = K[0] * p0 + K[1] * p1 + K[2] * p2;
        float q1 = K[3] * p0 + K[4] * p1 + K[5] * p2;
        float q2 = K[6] * p0 + K[7] * p1 + K[8] * p2;
        float inv = 1.0f / (q2 + 1e-6f);
        float u = q0 * inv, w = q1 * inv;
#pragma unroll
        for (int s = 0; s < 3; ++s) {
            float uf = u / Ss[s], vf = w / Ss[s];
            bool valid = (p2 > 0.1f) && (uf >= 0.f) && (uf < (float)Ws[s]) &&
                         (vf >= 0.f) && (vf < (float)Hs[s]);
            if (valid && !found[s]) {
                found[s] = 1;
                flat[s]  = cam * (Hs[s] * Ws[s]) + (int)vf * Ws[s] + (int)uf;
            }
        }
    }
    lut[0 * NV + v] = flat[0];
    lut[1 * NV + v] = found[0];
    lut[2 * NV + v] = flat[1];
    lut[3 * NV + v] = found[1];
    lut[4 * NV + v] = flat[2];
    lut[5 * NV + v] = found[2];
}

// ---------------------------------------------------------------------------
// LUT gather -> bev f32 (d_out) and bev_h f16 NHWC for the WMMA encoder.
// ---------------------------------------------------------------------------
__global__ void gather_kernel(const _Float16* __restrict__ f1,
                              const _Float16* __restrict__ f2,
                              const _Float16* __restrict__ f3,
                              const int* __restrict__ lut,
                              float* __restrict__ bev,
                              _Float16* __restrict__ bev_h)
{
    const int NV = 128 * 128 * 6;
    long idx = (long)blockIdx.x * blockDim.x + threadIdx.x;
    const long total = (long)NV * 224;
    if (idx >= total) return;
    int v = (int)(idx / 224);
    int c = (int)(idx - (long)v * 224);

    const _Float16* feat; int Cc, cl; const int* fl; const int* mk;
    if (c < 32)      { feat = f1; Cc = 32;  cl = c;      fl = lut;          mk = lut + NV; }
    else if (c < 96) { feat = f2; Cc = 64;  cl = c - 32; fl = lut + 2 * NV; mk = lut + 3 * NV; }
    else             { feat = f3; Cc = 128; cl = c - 96; fl = lut + 4 * NV; mk = lut + 5 * NV; }

    float val = 0.f;
    if (mk[v]) val = (float)feat[(size_t)fl[v] * Cc + cl];

    int iz = v % 6;
    int t  = v / 6;
    int iy = t % 128;
    int ix = t / 128;
    int bc = c * 6 + iz;
    bev[(size_t)bc * 16384 + ix * 128 + iy] = val;
    bev_h[((size_t)ix * 128 + iy) * 1344 + bc] = (_Float16)val;
}

// ---------------------------------------------------------------------------
// 1x1 head: 128 -> 10, f32 out (NCHW).
// ---------------------------------------------------------------------------
__global__ void head_kernel(const _Float16* __restrict__ h,
                            const float* __restrict__ w,
                            const float* __restrict__ b,
                            float* __restrict__ preds)
{
    int p = blockIdx.x * blockDim.x + threadIdx.x;
    if (p >= 16384) return;
    float acc[10];
#pragma unroll
    for (int o = 0; o < 10; ++o) acc[o] = b[o];
    const _Float16* hp = h + (size_t)p * 128;
    for (int c = 0; c < 128; ++c) {
        float hv = (float)hp[c];
#pragma unroll
        for (int o = 0; o < 10; ++o) acc[o] += hv * w[o * 128 + c];
    }
#pragma unroll
    for (int o = 0; o < 10; ++o) preds[(size_t)o * 16384 + p] = acc[o];
}

// ---------------------------------------------------------------------------
// Host launcher
// ---------------------------------------------------------------------------
extern "C" void kernel_launch(void* const* d_in, const int* in_sizes, int n_in,
                              void* d_out, int out_size, void* d_ws, size_t ws_size,
                              hipStream_t stream)
{
    (void)in_sizes; (void)n_in; (void)out_size; (void)ws_size;

    const float* imgs   = (const float*)d_in[0];
    const float* intr   = (const float*)d_in[1];
    const float* extr   = (const float*)d_in[2];
    const float* w_s1c1 = (const float*)d_in[3];
    const float* b_s1c1 = (const float*)d_in[4];
    const float* w_s1c2 = (const float*)d_in[5];
    const float* b_s1c2 = (const float*)d_in[6];
    const float* w_s2   = (const float*)d_in[7];
    const float* b_s2   = (const float*)d_in[8];
    const float* w_s3   = (const float*)d_in[9];
    const float* b_s3   = (const float*)d_in[10];
    const float* w_e1   = (const float*)d_in[11];
    const float* b_e1   = (const float*)d_in[12];
    const float* w_e2   = (const float*)d_in[13];
    const float* b_e2   = (const float*)d_in[14];
    const float* w_head = (const float*)d_in[15];
    const float* b_head = (const float*)d_in[16];
    const float* bn1_g = (const float*)d_in[17], *bn1_b = (const float*)d_in[18];
    const float* bn1_m = (const float*)d_in[19], *bn1_v = (const float*)d_in[20];
    const float* bn2_g = (const float*)d_in[21], *bn2_b = (const float*)d_in[22];
    const float* bn2_m = (const float*)d_in[23], *bn2_v = (const float*)d_in[24];
    const float* bn3_g = (const float*)d_in[25], *bn3_b = (const float*)d_in[26];
    const float* bn3_m = (const float*)d_in[27], *bn3_v = (const float*)d_in[28];
    const float* e1_g  = (const float*)d_in[29], *e1_b = (const float*)d_in[30];
    const float* e1_m  = (const float*)d_in[31], *e1_v = (const float*)d_in[32];
    const float* e2_g  = (const float*)d_in[33], *e2_b = (const float*)d_in[34];
    const float* e2_m  = (const float*)d_in[35], *e2_v = (const float*)d_in[36];

    char* wsb = (char*)d_ws;
    size_t off = 0;
    auto alloc = [&](size_t bytes) -> void* {
        void* p = wsb + off;
        off += (bytes + 255) & ~(size_t)255;
        return p;
    };
    _Float16* wt_s1c2 = (_Float16*)alloc((size_t)9 * 32  * 32   * 2);
    _Float16* wt_s2   = (_Float16*)alloc((size_t)9 * 64  * 32   * 2);
    _Float16* wt_s3   = (_Float16*)alloc((size_t)9 * 128 * 64   * 2);
    _Float16* wt_e1   = (_Float16*)alloc((size_t)9 * 128 * 1344 * 2);
    _Float16* wt_e2   = (_Float16*)alloc((size_t)9 * 128 * 128  * 2);
    float* sb_s1c2 = (float*)alloc((size_t)2 * 32  * 4);
    float* sb_s2   = (float*)alloc((size_t)2 * 64  * 4);
    float* sb_s3   = (float*)alloc((size_t)2 * 128 * 4);
    float* sb_e1   = (float*)alloc((size_t)2 * 128 * 4);
    float* sb_e2   = (float*)alloc((size_t)2 * 128 * 4);
    _Float16* zbuf = (_Float16*)alloc((size_t)256);
    _Float16* out0 = (_Float16*)alloc((size_t)6 * 128 * 352 * 32  * 2);
    _Float16* f1   = (_Float16*)alloc((size_t)6 * 64  * 176 * 32  * 2);
    _Float16* f2   = (_Float16*)alloc((size_t)6 * 32  * 88  * 64  * 2);
    _Float16* f3   = (_Float16*)alloc((size_t)6 * 16  * 44  * 128 * 2);
    int* lut       = (int*)alloc((size_t)6 * 98304 * 4);
    _Float16* bev_h = (_Float16*)alloc((size_t)16384 * 1344 * 2);
    _Float16* h1    = (_Float16*)alloc((size_t)16384 * 128  * 2);
    _Float16* h2    = (_Float16*)alloc((size_t)16384 * 128  * 2);

    float* preds = (float*)d_out;
    float* bev   = (float*)d_out + 163840;

    // --- parameter prep (tiny) ---
    zerofill_kernel<<<1, 128, 0, stream>>>(zbuf, 128);
    bnfold_kernel<<<1, 128, 0, stream>>>(b_s1c2, bn1_g, bn1_b, bn1_m, bn1_v, sb_s1c2, sb_s1c2 + 32, 32);
    bnfold_kernel<<<1, 128, 0, stream>>>(b_s2,   bn2_g, bn2_b, bn2_m, bn2_v, sb_s2,   sb_s2 + 64,   64);
    bnfold_kernel<<<1, 128, 0, stream>>>(b_s3,   bn3_g, bn3_b, bn3_m, bn3_v, sb_s3,   sb_s3 + 128, 128);
    bnfold_kernel<<<1, 128, 0, stream>>>(b_e1,   e1_g,  e1_b,  e1_m,  e1_v,  sb_e1,   sb_e1 + 128, 128);
    bnfold_kernel<<<1, 128, 0, stream>>>(b_e2,   e2_g,  e2_b,  e2_m,  e2_v,  sb_e2,   sb_e2 + 128, 128);

    auto wx = [&](const float* w, _Float16* wt, int Cout, int Cin) {
        int total = 9 * Cout * Cin;
        wxform_kernel<<<(total + 255) / 256, 256, 0, stream>>>(w, wt, Cout, Cin);
    };
    wx(w_s1c2, wt_s1c2, 32, 32);
    wx(w_s2,   wt_s2,   64, 32);
    wx(w_s3,   wt_s3,   128, 64);
    wx(w_e1,   wt_e1,   128, 1344);
    wx(w_e2,   wt_e2,   128, 128);

    // --- backbone ---
    {
        int npix = 6 * 128 * 352;
        conv1_direct<<<(npix + 255) / 256, 256, 0, stream>>>(imgs, w_s1c1, b_s1c1, out0);
    }
    auto convS = [&](const _Float16* in, const _Float16* wt, const float* sb,
                     _Float16* out, int Nimg, int Hin, int Win, int Cin,
                     int Hout, int Wout, int Cout, int stride, int biasOff) {
        int tiles = ((Nimg * Hout * Wout + 15) / 16) * (Cout / 16);
        conv3x3_wmma<<<(tiles + 7) / 8, 256, 0, stream>>>(
            in, wt, sb, sb + biasOff, zbuf, out,
            Nimg, Hin, Win, Cin, Hout, Wout, Cout, stride);
    };
    auto convL = [&](const _Float16* in, const _Float16* wt, const float* sb,
                     _Float16* out, int Nimg, int Hin, int Win, int Cin,
                     int Hout, int Wout, int stride) {
        int blocks = (Nimg * Hout * Wout) / 64;     // Cout fixed at 128
        conv3x3_wmma_lds<<<blocks, 256, 0, stream>>>(
            in, wt, sb, sb + 128, zbuf, out,
            Nimg, Hin, Win, Cin, Hout, Wout, stride);
    };
    convS(out0, wt_s1c2, sb_s1c2, f1, 6, 128, 352, 32, 64, 176, 32, 2, 32);
    convS(f1,   wt_s2,   sb_s2,   f2, 6, 64,  176, 32, 32, 88,  64, 2, 64);
    convL(f2,   wt_s3,   sb_s3,   f3, 6, 32,  88,  64, 16, 44, 2);

    // --- projection + gather ---
    lut_kernel<<<(98304 + 255) / 256, 256, 0, stream>>>(intr, extr, lut);
    {
        long total = 98304L * 224;
        gather_kernel<<<(unsigned)((total + 255) / 256), 256, 0, stream>>>(
            f1, f2, f3, lut, bev, bev_h);
    }

    // --- BEV encoder (dominant compute: 50.7 GFLOP in e1) ---
    convL(bev_h, wt_e1, sb_e1, h1, 1, 128, 128, 1344, 128, 128, 1);
    convL(h1,    wt_e2, sb_e2, h2, 1, 128, 128, 128,  128, 128, 1);

    // --- head ---
    head_kernel<<<64, 256, 0, stream>>>(h2, w_head, b_head, preds);
}
